// Block_56126632624726
// MI455X (gfx1250) — matrix operand order repaired and checked by reading, not compiled, not run on hardware
//
#include <hip/hip_runtime.h>
#include <math.h>

// ---------------------------------------------------------------------------
// MI455X (gfx1250, wave32) fused transformer block.
//   FLOPs ~190G, HBM traffic ~250MB -> compute bound; use bf16 WMMA
//   (v_wmma_f32_16x16x32_bf16) with f32 accumulation for all matmuls,
//   flash-style causal power-softmax (p=2) attention, f32 statistics.
// GEMMs: 64x32 C-tile per wave (8 WMMA / 12 b128 loads per 32-K step),
// compile-time K -> fully unrolled loop, immediate-offset addressing.
// ---------------------------------------------------------------------------

typedef __attribute__((ext_vector_type(16))) __bf16 v16bf;
typedef __attribute__((ext_vector_type(8)))  __bf16 v8bf;
typedef __attribute__((ext_vector_type(8)))  float  v8f;

constexpr int kDim   = 1024;
constexpr int kB     = 4;
constexpr int kL     = 2048;
constexpr int kRows  = kB * kL;     // 8192
constexpr int kNProj = 5 * kDim;    // 5120
constexpr int kHeads = 8;
constexpr int kHd    = 128;         // head dim
constexpr float kEps = 1e-5f;

#define WMMA_BF16(a, b, c) \
  __builtin_amdgcn_wmma_f32_16x16x32_bf16(false, (a), false, (b), (short)0, (c), false, false)

union AFrag { v16bf v; v8bf h[2]; };

__device__ __forceinline__ float bf2f(__bf16 b) {
  unsigned short u = __builtin_bit_cast(unsigned short, b);
  unsigned x = ((unsigned)u) << 16;
  return __builtin_bit_cast(float, x);
}
__device__ __forceinline__ __bf16 f2bf(float f) {
  unsigned x = __builtin_bit_cast(unsigned, f);
  unsigned r = x + 0x7FFFu + ((x >> 16) & 1u);        // round-to-nearest-even
  if ((x & 0x7F800000u) == 0x7F800000u) r = x;        // inf/nan passthrough
  return __builtin_bit_cast(__bf16, (unsigned short)(r >> 16));
}

// A fragment (16x32 bf16) from row-major [M,K] storage.
// lane<16: row m0+lane,  K = k0+{0..7, 16..23}
// lane>=16: row m0+lane-16, K = k0+{8..15, 24..31}
__device__ __forceinline__ v16bf load_a_frag(const __bf16* base, int lda, int m0, int k0, int lane) {
  int half = lane >> 4;
  int row  = m0 + (lane & 15);
  const __bf16* p = base + (size_t)row * lda + k0 + half * 8;
  AFrag f;
  f.h[0] = *(const v8bf*)(p);
  f.h[1] = *(const v8bf*)(p + 16);
  return f.v;
}

// B fragment (32x16 bf16) for B = W^T with W stored row-major [N,K].
// lane<16: col n0+lane,  K = k0+{0..7, 16..23} ; lane>=16: col n0+lane-16, K = k0+{8..15, 24..31}
__device__ __forceinline__ v16bf load_b_frag(const __bf16* base, int ldb, int n0, int k0, int lane) {
  int half = lane >> 4;
  int col  = n0 + (lane & 15);
  const __bf16* p = base + (size_t)col * ldb + k0 + half * 8;
  AFrag f;
  f.h[0] = *(const v8bf*)(p);
  f.h[1] = *(const v8bf*)(p + 16);
  return f.v;
}

// ---------------------------------------------------------------------------
// K1: LayerNorm (f32 stats) + cast to bf16.  One block per row of 1024.
// ---------------------------------------------------------------------------
__global__ void __launch_bounds__(256) ln_cast_kernel(const float* __restrict__ x,
                                                      const float* __restrict__ gamma,
                                                      const float* __restrict__ beta,
                                                      __bf16* __restrict__ h) {
  int row = blockIdx.x, t = threadIdx.x;
  const float4* xr = (const float4*)(x + (size_t)row * kDim);
  float4 v = xr[t];
  float s  = v.x + v.y + v.z + v.w;
  float s2 = v.x * v.x + v.y * v.y + v.z * v.z + v.w * v.w;
#pragma unroll
  for (int off = 16; off >= 1; off >>= 1) {
    s  += __shfl_xor(s,  off, 32);
    s2 += __shfl_xor(s2, off, 32);
  }
  __shared__ float rs[8], rs2[8];
  if ((t & 31) == 0) { rs[t >> 5] = s; rs2[t >> 5] = s2; }
  __syncthreads();
  float S = 0.f, S2 = 0.f;
#pragma unroll
  for (int i = 0; i < 8; ++i) { S += rs[i]; S2 += rs2[i]; }
  float mu   = S * (1.0f / kDim);
  float var  = S2 * (1.0f / kDim) - mu * mu;
  float rstd = rsqrtf(var + kEps);
  int c = t * 4;
  float4 g  = ((const float4*)gamma)[t];
  float4 bb = ((const float4*)beta)[t];
  __bf16* hr = h + (size_t)row * kDim;
  hr[c + 0] = f2bf((v.x - mu) * rstd * g.x + bb.x);
  hr[c + 1] = f2bf((v.y - mu) * rstd * g.y + bb.y);
  hr[c + 2] = f2bf((v.z - mu) * rstd * g.z + bb.z);
  hr[c + 3] = f2bf((v.w - mu) * rstd * g.w + bb.w);
}

// ---------------------------------------------------------------------------
// K2: f32 -> bf16 weight cast.
// ---------------------------------------------------------------------------
__global__ void __launch_bounds__(256) cast_f32_bf16_kernel(const float* __restrict__ in,
                                                            __bf16* __restrict__ out, int n) {
  int i = blockIdx.x * 256 + threadIdx.x;
  if (i < n) out[i] = f2bf(in[i]);
}

// ---------------------------------------------------------------------------
// GEMM core: C[M,N] = A[M,K] * W[N,K]^T.
// Block = 8 waves (256 thr) tiled 2(m) x 4(n): block tile 128x128.
// Wave tile 64x32: 8 WMMA + 12 b128 loads per 32-K step.
// K compile-time -> full unroll, all loads base-pointer + immediate offset.
// ---------------------------------------------------------------------------
template <int K, bool RESIDUAL>
__device__ __forceinline__ void gemm_body(const __bf16* __restrict__ A,
                                          const __bf16* __restrict__ W,
                                          __bf16* __restrict__ Cbf,
                                          const float* __restrict__ x,
                                          float* __restrict__ Cf32,
                                          int N) {
  int lane = threadIdx.x & 31;
  int wave = threadIdx.x >> 5;
  int wm = wave & 1, wn = wave >> 1;
  int m0 = blockIdx.x * 128 + wm * 64;
  int n0 = blockIdx.y * 128 + wn * 32;
  int ln = lane & 15, half = lane >> 4;

  // per-lane base pointers (fixed; K-loop uses immediate offsets only)
  const __bf16* ap[4];
  const __bf16* bp[2];
#pragma unroll
  for (int i = 0; i < 4; ++i) ap[i] = A + (size_t)(m0 + i * 16 + ln) * K + half * 8;
#pragma unroll
  for (int j = 0; j < 2; ++j) bp[j] = W + (size_t)(n0 + j * 16 + ln) * K + half * 8;

  v8f acc[4][2];
#pragma unroll
  for (int i = 0; i < 4; ++i)
#pragma unroll
    for (int j = 0; j < 2; ++j) acc[i][j] = (v8f){};

#pragma unroll
  for (int k = 0; k < K; k += 32) {
    v16bf b[2], a[4];
#pragma unroll
    for (int j = 0; j < 2; ++j) {
      AFrag f;
      f.h[0] = *(const v8bf*)(bp[j] + k);
      f.h[1] = *(const v8bf*)(bp[j] + k + 16);
      b[j] = f.v;
    }
#pragma unroll
    for (int i = 0; i < 4; ++i) {
      AFrag f;
      f.h[0] = *(const v8bf*)(ap[i] + k);
      f.h[1] = *(const v8bf*)(ap[i] + k + 16);
      a[i] = f.v;
    }
#pragma unroll
    for (int i = 0; i < 4; ++i)
#pragma unroll
      for (int j = 0; j < 2; ++j) acc[i][j] = WMMA_BF16(a[i], b[j], acc[i][j]);
  }

  int rowoff = half * 8;
#pragma unroll
  for (int i = 0; i < 4; ++i) {
#pragma unroll
    for (int r = 0; r < 8; ++r) {
      size_t rr = (size_t)(m0 + i * 16 + r + rowoff) * N + n0;
#pragma unroll
      for (int j = 0; j < 2; ++j) {
        size_t idx = rr + j * 16 + ln;
        if (RESIDUAL) Cf32[idx] = x[idx] + acc[i][j][r];
        else          Cbf[idx]  = f2bf(acc[i][j][r]);
      }
    }
  }
}

__global__ void __launch_bounds__(256) gemm_qkv_kernel(const __bf16* __restrict__ A,
                                                       const __bf16* __restrict__ W,
                                                       __bf16* __restrict__ C, int N) {
  gemm_body<kDim, false>(A, W, C, nullptr, nullptr, N);
}

__global__ void __launch_bounds__(256) gemm_out_kernel(const __bf16* __restrict__ A,
                                                       const __bf16* __restrict__ W,
                                                       const float* __restrict__ x,
                                                       float* __restrict__ out, int N) {
  gemm_body<2 * kDim, true>(A, W, nullptr, x, out, N);
}

// ---------------------------------------------------------------------------
// K4: branch[:, 0:1024] = lin * gelu(pre)   (exact gelu, f32 math)
// ---------------------------------------------------------------------------
__global__ void __launch_bounds__(256) glu_kernel(const __bf16* __restrict__ proj,
                                                  __bf16* __restrict__ branch) {
  size_t i = (size_t)blockIdx.x * 256 + threadIdx.x;
  int row = (int)(i >> 10), col = (int)(i & 1023);
  const __bf16* pr = proj + (size_t)row * kNProj;
  float lin = bf2f(pr[3 * kDim + col]);
  float pre = bf2f(pr[4 * kDim + col]);
  float g = 0.5f * pre * (1.0f + erff(pre * 0.7071067811865475f));
  branch[(size_t)row * (2 * kDim) + col] = f2bf(lin * g);
}

// ---------------------------------------------------------------------------
// K5: V transpose: Vt[b,h][dim][key] = proj[b*L+key][2048 + h*128 + dim]
// Tiled 32x32 LDS transpose; coalesced both ways.
// ---------------------------------------------------------------------------
__global__ void __launch_bounds__(256) vtrans_kernel(const __bf16* __restrict__ proj,
                                                     __bf16* __restrict__ vt) {
  __shared__ __bf16 tile[32][33];
  int bh = blockIdx.z, bb = bh >> 3, hh = bh & 7;
  int k0 = blockIdx.x * 32, d0 = blockIdx.y * 32;
  int tx = threadIdx.x & 31, ty = threadIdx.x >> 5;
#pragma unroll
  for (int i = 0; i < 4; ++i) {
    int key = ty + i * 8;
    tile[key][tx] = proj[(size_t)(bb * kL + k0 + key) * kNProj + 2 * kDim + hh * kHd + d0 + tx];
  }
  __syncthreads();
#pragma unroll
  for (int i = 0; i < 4; ++i) {
    int dy = ty + i * 8;
    vt[((size_t)bh * kHd + d0 + dy) * kL + k0 + tx] = tile[tx][dy];
  }
}

// ---------------------------------------------------------------------------
// K6: flash attention with power-softmax p=2.
// One wave per (b, h, 16-query tile).  32-key chunks:
//   S = Q K^T (8 WMMAs), online m/d with d = sum exp(2(s-m)),
//   P relayout C->A via 1KB LDS, O += P V (8 WMMAs from pre-transposed Vt).
// Final: O * rsqrt(d)  -> exp(s - lse) with lse = logsumexp(2s)/2.
// ---------------------------------------------------------------------------
__global__ void __launch_bounds__(32) attn_kernel(const __bf16* __restrict__ proj,
                                                  const __bf16* __restrict__ vt,
                                                  __bf16* __restrict__ branch) {
  __shared__ __align__(16) __bf16 Pl[16 * 32];
  int lane = threadIdx.x;
  int bid = blockIdx.x;
  int qt = bid & 127;
  int bh = bid >> 7;          // b*HEADS + h
  int hh = bh & 7, bb = bh >> 3;
  int q0 = qt * 16;
  const __bf16* qbase = proj + (size_t)bb * kL * kNProj + hh * kHd;
  const __bf16* kb_   = proj + (size_t)bb * kL * kNProj + kDim + hh * kHd;
  const __bf16* vtb   = vt + (size_t)bh * kHd * kL;

  v16bf qa[4];
#pragma unroll
  for (int c = 0; c < 4; ++c) qa[c] = load_a_frag(qbase, kNProj, q0, c * 32, lane);

  float m[8], d[8];
  v8f o[8] = {};
#pragma unroll
  for (int r = 0; r < 8; ++r) { m[r] = -INFINITY; d[r] = 0.f; }

  int ln = lane & 15, halfsel = lane >> 4, rowoff = halfsel * 8;
  const float scale = 0.08838834764831845f;  // 1/sqrt(128)
  int nchunks = (q0 + 16 + 31) >> 5;         // causal: keys < q0+16

  for (int kb = 0; kb < nchunks; ++kb) {
    int kk = kb * 32;
    v8f s0 = {}, s1 = {};
#pragma unroll
    for (int c = 0; c < 4; ++c) {
      v16bf bk0 = load_b_frag(kb_, kNProj, kk,      c * 32, lane);
      v16bf bk1 = load_b_frag(kb_, kNProj, kk + 16, c * 32, lane);
      s0 = WMMA_BF16(qa[c], bk0, s0);
      s1 = WMMA_BF16(qa[c], bk1, s1);
    }
    // scale + causal mask + per-lane row max
    float lm[8];
#pragma unroll
    for (int r = 0; r < 8; ++r) {
      int rq = q0 + r + rowoff;
      float v0 = s0[r] * scale; if (kk + ln > rq)      v0 = -1e30f;
      float v1 = s1[r] * scale; if (kk + 16 + ln > rq) v1 = -1e30f;
      s0[r] = v0; s1[r] = v1;
      lm[r] = fmaxf(v0, v1);
    }
    // reduce max across the 16-lane half (C-layout row group)
#pragma unroll
    for (int off = 1; off < 16; off <<= 1)
#pragma unroll
      for (int r = 0; r < 8; ++r) lm[r] = fmaxf(lm[r], __shfl_xor(lm[r], off, 16));

    float alpha[8], ssum[8];
#pragma unroll
    for (int r = 0; r < 8; ++r) {
      float mn = fmaxf(m[r], lm[r]);
      alpha[r] = __expf(m[r] - mn);
      m[r] = mn;
      float p0 = __expf(s0[r] - mn), p1 = __expf(s1[r] - mn);
      s0[r] = p0; s1[r] = p1;
      ssum[r] = p0 * p0 + p1 * p1;   // p=2 power-softmax: d = sum exp(2(s-m))
    }
#pragma unroll
    for (int off = 1; off < 16; off <<= 1)
#pragma unroll
      for (int r = 0; r < 8; ++r) ssum[r] += __shfl_xor(ssum[r], off, 16);
#pragma unroll
    for (int r = 0; r < 8; ++r) d[r] = d[r] * alpha[r] * alpha[r] + ssum[r];
#pragma unroll
    for (int t = 0; t < 8; ++t)
#pragma unroll
      for (int r = 0; r < 8; ++r) o[t][r] *= alpha[r];

    // C-layout -> A-layout relayout of P through LDS
    __syncthreads();
#pragma unroll
    for (int r = 0; r < 8; ++r) {
      int row = r + rowoff;
      Pl[row * 32 + ln]      = f2bf(s0[r]);
      Pl[row * 32 + 16 + ln] = f2bf(s1[r]);
    }
    __syncthreads();
    AFrag pf;
    const __bf16* pp = Pl + ln * 32 + halfsel * 8;
    pf.h[0] = *(const v8bf*)(pp);
    pf.h[1] = *(const v8bf*)(pp + 16);

    // O += P * V  (B-fragments contiguous from pre-transposed Vt)
#pragma unroll
    for (int t = 0; t < 8; ++t) {
      v16bf vb = load_b_frag(vtb, kL, t * 16, kk, lane);
      o[t] = WMMA_BF16(pf.v, vb, o[t]);
    }
  }

  float rinv[8];
#pragma unroll
  for (int r = 0; r < 8; ++r) rinv[r] = rsqrtf(d[r]);
#pragma unroll
  for (int t = 0; t < 8; ++t)
#pragma unroll
    for (int r = 0; r < 8; ++r) {
      size_t row = (size_t)(bb * kL + q0 + r + rowoff);
      branch[row * (2 * kDim) + kDim + hh * kHd + t * 16 + ln] = f2bf(o[t][r] * rinv[r]);
    }
}

// ---------------------------------------------------------------------------
// Host launcher.  Workspace layout (bytes, all 4KB-aligned offsets):
//   h_bf    [8192,1024] bf16   @ 0          (16.8 MB)
//   wqkv_bf [5120,1024] bf16   @ 16777216   (10.5 MB)
//   wout_bf [1024,2048] bf16   @ 27262976   ( 4.2 MB)
//   proj_bf [8192,5120] bf16   @ 31457280   (83.9 MB)
//   vt_bf   [32,128,2048] bf16 @ 115343360  (16.8 MB)
//   br_bf   [8192,2048] bf16   @ 132120576  (33.6 MB)   total ~166 MB
// ---------------------------------------------------------------------------
extern "C" void kernel_launch(void* const* d_in, const int* in_sizes, int n_in,
                              void* d_out, int out_size, void* d_ws, size_t ws_size,
                              hipStream_t stream) {
  (void)in_sizes; (void)n_in; (void)out_size; (void)ws_size;
  const float* x     = (const float*)d_in[0];
  const float* gamma = (const float*)d_in[1];
  const float* beta  = (const float*)d_in[2];
  const float* wqkv  = (const float*)d_in[3];
  const float* wout  = (const float*)d_in[4];
  float* out = (float*)d_out;

  char* ws = (char*)d_ws;
  __bf16* h_bf    = (__bf16*)(ws);
  __bf16* wqkv_bf = (__bf16*)(ws + 16777216);
  __bf16* wout_bf = (__bf16*)(ws + 27262976);
  __bf16* proj_bf = (__bf16*)(ws + 31457280);
  __bf16* vt_bf   = (__bf16*)(ws + 115343360);
  __bf16* br_bf   = (__bf16*)(ws + 132120576);

  ln_cast_kernel<<<kRows, 256, 0, stream>>>(x, gamma, beta, h_bf);
  cast_f32_bf16_kernel<<<(5242880 + 255) / 256, 256, 0, stream>>>(wqkv, wqkv_bf, 5242880);
  cast_f32_bf16_kernel<<<(2097152 + 255) / 256, 256, 0, stream>>>(wout, wout_bf, 2097152);

  // proj = h @ w_qkv^T : M=8192, N=5120, K=1024  (block tile 128x128)
  gemm_qkv_kernel<<<dim3(kRows / 128, kNProj / 128), 256, 0, stream>>>(
      h_bf, wqkv_bf, proj_bf, kNProj);

  glu_kernel<<<(kRows * kDim) / 256, 256, 0, stream>>>(proj_bf, br_bf);
  vtrans_kernel<<<dim3(kL / 32, kHd / 32, kB * kHeads), 256, 0, stream>>>(proj_bf, vt_bf);
  attn_kernel<<<kB * kHeads * (kL / 16), 32, 0, stream>>>(proj_bf, vt_bf, br_bf);

  // out = x + branch @ w_out^T : M=8192, N=1024, K=2048
  gemm_out_kernel<<<dim3(kRows / 128, kDim / 128), 256, 0, stream>>>(
      br_bf, wout_bf, x, out, kDim);
}
